// Attention_90125593739547
// MI455X (gfx1250) — compile-verified, matrix operand order checked by reading, not run on hardware
//
#include <hip/hip_runtime.h>
#include <math.h>

// Shapes (fixed by the reference)
#define S_LEN 2048
#define BATCH 2
#define DIM   1024
#define NH    16
#define HD    64
#define MROWS (S_LEN * BATCH)          // 4096
#define QSCALE 0.180336880f            // (1/sqrt(64)) * log2(e), folded into Q

typedef __attribute__((ext_vector_type(8)))  __bf16 v8bf;
typedef __attribute__((ext_vector_type(16))) __bf16 v16bf;
typedef __attribute__((ext_vector_type(8)))  float  v8f;

// ---------------------------------------------------------------------------
// Async global -> LDS copy of one 16-byte chunk per lane (CDNA5 TDM-lite
// path: VDST = per-lane LDS byte offset, VADDR = 64-bit global address,
// tracked by ASYNCcnt). LDS offset = low 32 bits of the generic address
// (ISA: LDS aperture maps addr[31:0] directly to LDS).
// ---------------------------------------------------------------------------
__device__ __forceinline__ void async_load_b128(const __bf16* gptr,
                                                __bf16* lptr) {
  unsigned lds_off = (unsigned)(uintptr_t)lptr;
  unsigned long long ga = (unsigned long long)(uintptr_t)gptr;
  asm volatile("global_load_async_to_lds_b128 %0, %1, off"
               :: "v"(lds_off), "v"(ga)
               : "memory");
}
__device__ __forceinline__ void wait_async_all() {
  asm volatile("s_wait_asynccnt 0" ::: "memory");
}

// ---------------------------------------------------------------------------
// WMMA 16x32 bf16 fragment loaders (A layout; also the B operand when the
// source is "N-major, K-contiguous").
// Lane l (0..15):  row l,      K elems {0..7, 16..23}
// Lane l (16..31): row (l&15), K elems {8..15, 24..31}
// ---------------------------------------------------------------------------
__device__ __forceinline__ v16bf combine16(v8bf x, v8bf y) {
  return __builtin_shufflevector(x, y, 0, 1, 2, 3, 4, 5, 6, 7,
                                       8, 9, 10, 11, 12, 13, 14, 15);
}
// from global memory (per-lane gather; used only for the Q frag)
__device__ __forceinline__ v16bf load_frag_g(const __bf16* __restrict__ base,
                                             int ld, int row0, int k0) {
  int lane = threadIdx.x & 31;
  int l = lane & 15, hi8 = (lane & 16) >> 1;
  const __bf16* p = base + (size_t)(row0 + l) * ld + k0 + hi8;
  return combine16(*(const v8bf*)(p), *(const v8bf*)(p + 16));
}
// from LDS: p already points at (row, kbase + hi8); two ds_load_b128
__device__ __forceinline__ v16bf lds_frag(const __bf16* p) {
  return combine16(*(const v8bf*)(p), *(const v8bf*)(p + 16));
}

// ---------------------------------------------------------------------------
// f32 -> bf16 convert / transpose-convert
// ---------------------------------------------------------------------------
__global__ void cvt_bf16_kernel(const float* __restrict__ src,
                                __bf16* __restrict__ dst, int n) {
  int i = blockIdx.x * blockDim.x + threadIdx.x;
  int stride = gridDim.x * blockDim.x;
  for (; i < n; i += stride) dst[i] = (__bf16)src[i];
}
__global__ void transpose_cvt_kernel(const float* __restrict__ src,
                                     __bf16* __restrict__ dst, int K, int N) {
  int idx = blockIdx.x * blockDim.x + threadIdx.x;
  if (idx >= K * N) return;
  int k = idx / N, n = idx - k * N;
  dst[(size_t)n * K + k] = (__bf16)src[idx];
}

// ---------------------------------------------------------------------------
// LDS-staged WMMA GEMM: C[Mt,Nt] = A[Mt,Kt] * Wt[Nt,Kt]^T + bias.
// Block = 8 waves = 128x64 output tile. Per K-step of 128: async-stage
// A(128x128, 32KB) + B(64x128, 16KB) into LDS with coalesced 16B chunks,
// barrier, then each wave runs 4 inner K-steps of 4 WMMAs from ds_load frags.
// MODE 0: KV -> K[B,H,S,HD] bf16, V[B,H,HD,S] bf16 (transposed)
// MODE 1: Q  -> Q[B,H,S,HD] bf16, pre-scaled by QSCALE
// MODE 2: out projection -> f32 [M,N] (+bias)
// ---------------------------------------------------------------------------
template <int MODE, int Mt, int Nt, int Kt>
__global__ void __launch_bounds__(256)
gemm_bf16_kernel(const __bf16* __restrict__ A, const __bf16* __restrict__ Wt,
                 const float* __restrict__ bias, __bf16* __restrict__ outK,
                 __bf16* __restrict__ outV, float* __restrict__ outF) {
  __shared__ __align__(16) __bf16 Atile[128][128];  // 32 KB
  __shared__ __align__(16) __bf16 Btile[64][128];   // 16 KB

  const int tilesN = Nt / 64;
  int mt = blockIdx.x / tilesN, nt = blockIdx.x - mt * tilesN;
  int row0 = mt * 128, col0 = nt * 64;
  int tid = threadIdx.x;
  int wv = tid >> 5, lane = tid & 31, l = lane & 15, hi = lane >> 4;
  int hi8 = (lane & 16) >> 1;

  v8f acc[4];
#pragma unroll
  for (int i = 0; i < 4; ++i) acc[i] = (v8f){};

  for (int k0 = 0; k0 < Kt; k0 += 128) {
    // stage A: 2048 chunks of 16B, 8 per thread, coalesced along K
#pragma unroll
    for (int c = 0; c < 8; ++c) {
      int chunk = tid + 256 * c;
      int r = chunk >> 4, kc = (chunk & 15) * 8;
      async_load_b128(A + (size_t)(row0 + r) * Kt + k0 + kc, &Atile[r][kc]);
    }
    // stage B: 1024 chunks, 4 per thread
#pragma unroll
    for (int c = 0; c < 4; ++c) {
      int chunk = tid + 256 * c;
      int r = chunk >> 4, kc = (chunk & 15) * 8;
      async_load_b128(Wt + (size_t)(col0 + r) * Kt + k0 + kc, &Btile[r][kc]);
    }
    wait_async_all();
    __syncthreads();

#pragma unroll
    for (int ki = 0; ki < 4; ++ki) {
      v16bf a = lds_frag(&Atile[wv * 16 + l][ki * 32 + hi8]);
#pragma unroll
      for (int nb = 0; nb < 4; ++nb) {
        v16bf b = lds_frag(&Btile[nb * 16 + l][ki * 32 + hi8]);
        acc[nb] = __builtin_amdgcn_wmma_f32_16x16x32_bf16(
            false, a, false, b, (short)0, acc[nb], false, false);
      }
    }
    __syncthreads();
  }

  int rw0 = row0 + wv * 16;
#pragma unroll
  for (int nb = 0; nb < 4; ++nb) {
#pragma unroll
    for (int r = 0; r < 8; ++r) {
      int m   = rw0 + r + 8 * hi;            // row = s*BATCH + b
      int col = col0 + nb * 16 + l;
      float v = acc[nb][r] + bias[col];
      if (MODE == 0) {                        // fused KV
        int s = m >> 1, b = m & 1;
        if (col < NH * HD) {
          int h = col >> 6, d = col & 63;
          outK[((((size_t)b * NH + h) * S_LEN + s) << 6) + d] = (__bf16)v;
        } else {
          int c2 = col - NH * HD;
          int h = c2 >> 6, d = c2 & 63;
          outV[(((size_t)b * NH + h) * HD + d) * S_LEN + s] = (__bf16)v;
        }
      } else if (MODE == 1) {                 // Q, pre-scaled into exp2 domain
        int s = m >> 1, b = m & 1;
        int h = col >> 6, d = col & 63;
        outK[((((size_t)b * NH + h) * S_LEN + s) << 6) + d] =
            (__bf16)(v * QSCALE);
      } else {                                // final output, f32
        outF[(size_t)m * Nt + col] = v;
      }
    }
  }
}

// ---------------------------------------------------------------------------
// Flash attention. Block = 4 waves, all on the same (b,h); each wave owns 16
// query rows. Per 32-wide t-step the block cooperatively async-stages the
// shared K tile (32x64) and V tile (64x32, pre-transposed) into LDS; frags
// come from ds_load_b128. Scores are already in the exp2 domain (Q carries
// SCALE*log2e), so the online softmax uses raw v_exp (exp2).
// ---------------------------------------------------------------------------
__global__ void __launch_bounds__(128)
attn_kernel(const __bf16* __restrict__ Qb, const __bf16* __restrict__ Kb,
            const __bf16* __restrict__ Vt, __bf16* __restrict__ attn) {
  __shared__ __align__(16) __bf16 Ktile[32][64];    // 4 KB
  __shared__ __align__(16) __bf16 Vtile[64][32];    // 4 KB
  __shared__ __align__(16) __bf16 plds[4][16][32];  // per-wave P tile, 4 KB

  int tid = threadIdx.x;
  int waveIn = tid >> 5;
  int bh = blockIdx.x >> 5;                  // 32 blocks per (b,h)
  int qt = ((blockIdx.x & 31) << 2) + waveIn;
  int b = bh >> 4, h = bh & 15;
  int q0 = qt * 16;
  int lane = tid & 31, l = lane & 15, hi = lane >> 4;
  int hi8 = (lane & 16) >> 1;

  const __bf16* Qh = Qb + (size_t)bh * S_LEN * HD;
  const __bf16* Kh = Kb + (size_t)bh * S_LEN * HD;
  const __bf16* Vh = Vt + (size_t)bh * HD * S_LEN;

  v16bf qa0 = load_frag_g(Qh, HD, q0, 0);
  v16bf qa1 = load_frag_g(Qh, HD, q0, 32);

  v8f o[4];
#pragma unroll
  for (int i = 0; i < 4; ++i) o[i] = (v8f){};
  float mrun[8], lrun[8];
#pragma unroll
  for (int r = 0; r < 8; ++r) { mrun[r] = -INFINITY; lrun[r] = 0.0f; }

  for (int t0 = 0; t0 < S_LEN; t0 += 32) {
    // ---- cooperative staging: K 256 chunks + V 256 chunks, 2 each/thread --
#pragma unroll
    for (int c = 0; c < 2; ++c) {
      int chunk = tid + 128 * c;
      int kr = chunk >> 3, kc = (chunk & 7) * 8;
      async_load_b128(Kh + (size_t)(t0 + kr) * HD + kc, &Ktile[kr][kc]);
      int vr = chunk >> 2, vc = (chunk & 3) * 8;
      async_load_b128(Vh + (size_t)vr * S_LEN + t0 + vc, &Vtile[vr][vc]);
    }
    wait_async_all();
    __syncthreads();

    // ---- scores: S = Q(16x64) * K^T (two 16-col sub-tiles) ----
    v16bf kb00 = lds_frag(&Ktile[l][hi8]);
    v16bf kb01 = lds_frag(&Ktile[l][32 + hi8]);
    v16bf kb10 = lds_frag(&Ktile[16 + l][hi8]);
    v16bf kb11 = lds_frag(&Ktile[16 + l][32 + hi8]);
    v8f z = (v8f){};
    v8f s0 = __builtin_amdgcn_wmma_f32_16x16x32_bf16(false, qa0, false, kb00,
                                                     (short)0, z, false, false);
    s0 = __builtin_amdgcn_wmma_f32_16x16x32_bf16(false, qa1, false, kb01,
                                                 (short)0, s0, false, false);
    v8f s1 = __builtin_amdgcn_wmma_f32_16x16x32_bf16(false, qa0, false, kb10,
                                                     (short)0, z, false, false);
    s1 = __builtin_amdgcn_wmma_f32_16x16x32_bf16(false, qa1, false, kb11,
                                                 (short)0, s1, false, false);

    // ---- online softmax (exp2 domain); row r lives in one 16-lane half ----
    float p0[8], p1[8], alpha[8];
#pragma unroll
    for (int r = 0; r < 8; ++r) {
      float a = s0[r], c = s1[r];
      float mt = fmaxf(a, c);
      mt = fmaxf(mt, __shfl_xor(mt, 1, 32));
      mt = fmaxf(mt, __shfl_xor(mt, 2, 32));
      mt = fmaxf(mt, __shfl_xor(mt, 4, 32));
      mt = fmaxf(mt, __shfl_xor(mt, 8, 32));
      float mn = fmaxf(mrun[r], mt);
      alpha[r] = __builtin_amdgcn_exp2f(mrun[r] - mn);
      mrun[r]  = mn;
      p0[r] = __builtin_amdgcn_exp2f(a - mn);
      p1[r] = __builtin_amdgcn_exp2f(c - mn);
      float rs = p0[r] + p1[r];
      rs += __shfl_xor(rs, 1, 32);
      rs += __shfl_xor(rs, 2, 32);
      rs += __shfl_xor(rs, 4, 32);
      rs += __shfl_xor(rs, 8, 32);
      lrun[r] = lrun[r] * alpha[r] + rs;
    }
#pragma unroll
    for (int nb = 0; nb < 4; ++nb)
#pragma unroll
      for (int r = 0; r < 8; ++r) o[nb][r] *= alpha[r];

    // ---- C-layout -> A-layout relayout of P (within-wave LDS tile) ----
#pragma unroll
    for (int r = 0; r < 8; ++r) {
      plds[waveIn][r + 8 * hi][l]      = (__bf16)p0[r];
      plds[waveIn][r + 8 * hi][16 + l] = (__bf16)p1[r];
    }
    v16bf pa = lds_frag(&plds[waveIn][l][hi8]);

    // ---- O += P(16x32) * V(32x64) ----
#pragma unroll
    for (int nb = 0; nb < 4; ++nb) {
      v16bf vb = lds_frag(&Vtile[nb * 16 + l][hi8]);
      o[nb] = __builtin_amdgcn_wmma_f32_16x16x32_bf16(false, pa, false, vb,
                                                      (short)0, o[nb], false,
                                                      false);
    }
    __syncthreads();   // K/V (and plds) consumed; safe to restage
  }

  // ---- normalize, write attn_vec as [S, B, H*HD] bf16 ----
  float inv[8];
#pragma unroll
  for (int r = 0; r < 8; ++r) inv[r] = 1.0f / lrun[r];
#pragma unroll
  for (int nb = 0; nb < 4; ++nb)
#pragma unroll
    for (int r = 0; r < 8; ++r) {
      int s = q0 + r + 8 * hi;
      attn[(((size_t)s * BATCH + b) << 10) + h * HD + nb * 16 + l] =
          (__bf16)(o[nb][r] * inv[r]);
    }
}

// ---------------------------------------------------------------------------
extern "C" void kernel_launch(void* const* d_in, const int* in_sizes, int n_in,
                              void* d_out, int out_size, void* d_ws,
                              size_t ws_size, hipStream_t stream) {
  const float* inp = (const float*)d_in[0];   // [S,B,D]
  const float* Wkv = (const float*)d_in[1];   // [D, 2*H*HD]
  const float* bkv = (const float*)d_in[2];
  const float* Wq  = (const float*)d_in[3];   // [D, H*HD]
  const float* bq  = (const float*)d_in[4];
  const float* Wp  = (const float*)d_in[5];   // [H*HD, D]
  const float* bp  = (const float*)d_in[6];
  float* out = (float*)d_out;

  // workspace carve-up (48 MB total)
  char* w = (char*)d_ws;
  __bf16* X16  = (__bf16*)w; w += (size_t)MROWS * DIM * 2;             // 8 MB
  __bf16* Wkvt = (__bf16*)w; w += (size_t)2 * NH * HD * DIM * 2;       // 4 MB
  __bf16* Wqt  = (__bf16*)w; w += (size_t)NH * HD * DIM * 2;           // 2 MB
  __bf16* Wpt  = (__bf16*)w; w += (size_t)DIM * NH * HD * 2;           // 2 MB
  __bf16* Qb   = (__bf16*)w; w += (size_t)BATCH * NH * S_LEN * HD * 2; // 8 MB
  __bf16* Kb   = (__bf16*)w; w += (size_t)BATCH * NH * S_LEN * HD * 2; // 8 MB
  __bf16* Vtb  = (__bf16*)w; w += (size_t)BATCH * NH * HD * S_LEN * 2; // 8 MB
  __bf16* attnB = (__bf16*)w;                                          // 8 MB

  // 1) bf16 conversions (weights transposed so B-frags are K-contiguous)
  cvt_bf16_kernel<<<2048, 256, 0, stream>>>(inp, X16, MROWS * DIM);
  transpose_cvt_kernel<<<(DIM * 2 * NH * HD) / 256, 256, 0, stream>>>(
      Wkv, Wkvt, DIM, 2 * NH * HD);
  transpose_cvt_kernel<<<(DIM * NH * HD) / 256, 256, 0, stream>>>(Wq, Wqt, DIM,
                                                                  NH * HD);
  transpose_cvt_kernel<<<(NH * HD * DIM) / 256, 256, 0, stream>>>(Wp, Wpt,
                                                                  NH * HD, DIM);

  // 2) projections: fused KV (K row-major, V transposed), then scaled Q
  gemm_bf16_kernel<0, MROWS, 2 * NH * HD, DIM>
      <<<(MROWS / 128) * (2 * NH * HD / 64), 256, 0, stream>>>(
          X16, Wkvt, bkv, Kb, Vtb, nullptr);
  gemm_bf16_kernel<1, MROWS, NH * HD, DIM>
      <<<(MROWS / 128) * (NH * HD / 64), 256, 0, stream>>>(X16, Wqt, bq, Qb,
                                                           nullptr, nullptr);

  // 3) flash attention: 1024 blocks x 4 waves (one 16-row q tile per wave)
  attn_kernel<<<1024, 128, 0, stream>>>(Qb, Kb, Vtb, attnB);

  // 4) output projection straight into d_out (f32)
  gemm_bf16_kernel<2, MROWS, DIM, NH * HD>
      <<<(MROWS / 128) * (DIM / 64), 256, 0, stream>>>(attnB, Wpt, bp, nullptr,
                                                       nullptr, out);
}
